// CoAttnGPBlock_17351667876070
// MI455X (gfx1250) — matrix operand order repaired and checked by reading, not compiled, not run on hardware
//
#include <hip/hip_runtime.h>
#include <math.h>

// ---------------------------------------------------------------------------
// CoAttnGPBlock for MI455X (gfx1250): convs as implicit-GEMM bf16 WMMA
// (v_wmma_f32_16x16x32_bf16, f32 accumulate), attention/MLP on VALU.
// Conv blocks stage a 3x66x64 input slab in LDS once (pixel-major, channel-
// contiguous, padded stride) so all 18 K-chunks read B fragments as aligned
// ds_read_b128 with no per-iteration global traffic or predication.
// ---------------------------------------------------------------------------

typedef __bf16 bf16_t;
typedef bf16_t v16bf __attribute__((ext_vector_type(16)));
typedef float  v8f   __attribute__((ext_vector_type(8)));

constexpr int B_  = 2;
constexpr int Cin = 64;
constexpr int C_  = 64;
constexpr int H_  = 256;
constexpr int W_  = 1216;
constexpr int Ns_ = 20000;
constexpr int K_  = 9;
constexpr int HW_ = H_ * W_;          // 311296
constexpr int D_  = 2 * C_ + 3;       // 131
constexpr int Dh_ = 65;               // MLP hidden
constexpr int KDIM = Cin * 9;         // 576 (GEMM K), ordered kpos*64 + ci
constexpr int WTILE = 64;             // pixels per block (contiguous in W)
constexpr int NBLK  = W_ / WTILE;     // 19
constexpr int PXS   = 72;             // padded channel-slots per pixel in LDS

// --------------------------- weight reorder + bf16 -------------------------
// Wb[set][c][kpos*64+ci] = w[set][c][ci][kh][kw], kpos = kh*3+kw
__global__ void prep_weights_kernel(const float* __restrict__ w0, const float* __restrict__ w1,
                                    const float* __restrict__ w2, const float* __restrict__ w3,
                                    const float* __restrict__ w4, const float* __restrict__ w5,
                                    bf16_t* __restrict__ Wb) {
  int i = blockIdx.x * blockDim.x + threadIdx.x;
  const int per = C_ * KDIM;          // 36864
  if (i >= 6 * per) return;
  int set = i / per, r = i - set * per;
  int c = r / KDIM, k = r - c * KDIM;
  int kpos = k >> 6, ci = k & 63;
  const float* w;
  switch (set) { case 0: w = w0; break; case 1: w = w1; break; case 2: w = w2; break;
                 case 3: w = w3; break; case 4: w = w4; break; default: w = w5; break; }
  Wb[(size_t)set * per + (size_t)c * KDIM + k] = (bf16_t)w[c * KDIM + ci * 9 + kpos];
}

// ---------------- transpose MLP first-layer weights (f32) ------------------
// w1T[j][i] = w1[i][j]  (each MLP thread then streams its row contiguously)
__global__ void prep_w1_kernel(const float* __restrict__ dw1, const float* __restrict__ rw1,
                               float* __restrict__ dw1T, float* __restrict__ rw1T) {
  int i = blockIdx.x * blockDim.x + threadIdx.x;
  if (i >= D_ * Dh_) return;
  int r = i / Dh_, j = i - r * Dh_;
  dw1T[j * D_ + r] = dw1[i];
  rw1T[j * D_ + r] = rw1[i];
}

// --------------------------- f32 -> bf16 convert ---------------------------
__global__ void cvt_kernel(const float* __restrict__ src, bf16_t* __restrict__ dst, long n) {
  long i = (long)blockIdx.x * blockDim.x + threadIdx.x;
  if (i < n) dst[i] = (bf16_t)src[i];
}

// ---------------------- implicit-GEMM 3x3 SAME conv ------------------------
// Block: 256 threads = 8 waves; tile = 64 out-channels x 64 pixels (one h row).
// GEMM: M=64 (out ch), N=64 (pixels), K=576; K-chunk=32 stays in one tap.
__global__ __launch_bounds__(256)
void conv3x3_wmma(const bf16_t* __restrict__ in, const bf16_t* __restrict__ Wk,
                  const float* __restrict__ bias, float* __restrict__ out_f32,
                  bf16_t* __restrict__ out_bf, const float* __restrict__ resid,
                  int do_relu)
{
  __shared__ bf16_t Ilds[3 * 66 * PXS];   // input slab [row][px][ci], 28512 B
  __shared__ bf16_t Abuf[2][64 * 32];     // weight K-chunk double buffer, 8 KB

  int blk  = blockIdx.x;
  int wblk = blk % NBLK;
  int h    = (blk / NBLK) % H_;
  int b    = blk / (NBLK * H_);
  int w0   = wblk * WTILE;

  const bf16_t* inb = in + (size_t)b * Cin * HW_;

  int t    = threadIdx.x;
  int lane = t & 31;
  int wv   = t >> 5;                  // wave 0..7
  int mb   = wv & 3;                  // M tile (16 ch)
  int nh   = wv >> 2;                 // N tile pair: nh*2, nh*2+1
  int l15  = lane & 15;
  int lhi  = lane >> 4;               // 0/1

  // ---- one-time input slab stage: coalesced row reads, transposed LDS ----
  // 192 (row, ci) tasks split across 8 waves; lanes cover pixels.
  for (int task = wv * 24; task < wv * 24 + 24; ++task) {
    int row = task >> 6;              // 0..2
    int ci  = task & 63;
    int py  = h + row - 1;
    bool rowok = (py >= 0) && (py < H_);
    const bf16_t* srcRow = inb + (size_t)ci * HW_ + (size_t)py * W_;
    for (int px = lane; px < 66; px += 32) {
      int gx = w0 - 1 + px;
      bf16_t v = (bf16_t)0.0f;
      if (rowok && gx >= 0 && gx < W_) v = srcRow[gx];
      Ilds[(row * 66 + px) * PXS + ci] = v;
    }
    if (rowok) __builtin_prefetch(srcRow + W_, 0, 1);  // warm L2 for h+1 blocks
  }
  // ---- prestage A chunk 0 (kpos=0, ci0=0) ----
  {
    int base = t << 3;                // 0..2047
    int m = base >> 5, k0 = base & 31;
    const bf16_t* src = Wk + (size_t)m * KDIM + k0;
    #pragma unroll
    for (int u = 0; u < 8; ++u) Abuf[0][base + u] = src[u];
  }

  v8f acc0 = {};
  v8f acc1 = {};
  int abase = ((mb << 4) + l15) * 32 + (lhi << 4);
  int kb    = lhi << 4;
  int j0    = (nh << 5) + l15;

  #pragma unroll 2
  for (int it = 0; it < 18; ++it) {
    __syncthreads();                  // chunk it staged; prev reads of other buf done
    if (it < 17) {                    // overlap: stage next A chunk
      int cidx = it + 1;
      int kpos = cidx >> 1, ci0 = (cidx & 1) << 5;
      int base = t << 3;
      int m = base >> 5, k0 = base & 31;
      const bf16_t* src = Wk + (size_t)m * KDIM + kpos * 64 + ci0 + k0;
      bf16_t* dst = &Abuf[cidx & 1][base];
      #pragma unroll
      for (int u = 0; u < 8; ++u) dst[u] = src[u];
    }
    int kpos = it >> 1, ci0 = (it & 1) << 5;
    int kh = kpos / 3, kw = kpos - kh * 3;
    const bf16_t* ab = &Abuf[it & 1][0];

    // fragments: lane half lhi holds k = 16*lhi + e in both A and B (contiguous k order)
    v16bf a, bb0, bb1;
    #pragma unroll
    for (int e = 0; e < 16; ++e)
      a[e] = ab[abase + e];
    int bo0 = (kh * 66 + j0 + kw) * PXS + ci0 + kb;   // 16 contiguous ch, 16B aligned
    int bo1 = bo0 + 16 * PXS;
    #pragma unroll
    for (int e = 0; e < 16; ++e) {
      bb0[e] = Ilds[bo0 + e];
      bb1[e] = Ilds[bo1 + e];
    }
    acc0 = __builtin_amdgcn_wmma_f32_16x16x32_bf16(false, a, false, bb0, (short)0, acc0, false, false);
    acc1 = __builtin_amdgcn_wmma_f32_16x16x32_bf16(false, a, false, bb1, (short)0, acc1, false, false);
  }

  // epilogue: bias (+residual) (+relu), f32 or bf16 store
  size_t outb = (size_t)b * C_ * HW_;
  int cb   = (mb << 4) + (lhi << 3);
  int col0 = w0 + (nh << 5) + l15;
  #pragma unroll
  for (int v = 0; v < 8; ++v) {
    int ch = cb + v;
    float bv = bias[ch];
    size_t o0 = outb + (size_t)ch * HW_ + (size_t)h * W_ + col0;
    size_t o1 = o0 + 16;
    float y0 = acc0[v] + bv;
    float y1 = acc1[v] + bv;
    if (resid)   { y0 += resid[o0]; y1 += resid[o1]; }
    if (do_relu) { y0 = fmaxf(y0, 0.f); y1 = fmaxf(y1, 0.f); }
    if (out_f32) { out_f32[o0] = y0; out_f32[o1] = y1; }
    if (out_bf)  { out_bf[o0] = (bf16_t)y0; out_bf[o1] = (bf16_t)y1; }
  }
}

// ------------------------- attention (gather+MLP) --------------------------
// One 64-thread block per (b, n). Faithful to reference: r_nn gathered from d0.
__global__ __launch_bounds__(64)
void attn_kernel(const bf16_t* __restrict__ d0, const bf16_t* __restrict__ r0,
                 const int* __restrict__ pc_idx, const int* __restrict__ nbrs_idx,
                 const float* __restrict__ nbrs_disp,
                 const float* __restrict__ dw1T, const float* __restrict__ db1,
                 const float* __restrict__ dw2, const float* __restrict__ db2,
                 const float* __restrict__ rw1T, const float* __restrict__ rb1,
                 const float* __restrict__ rw2, const float* __restrict__ rb2,
                 const float* __restrict__ d_bias, const float* __restrict__ r_bias,
                 float* __restrict__ d_feat, float* __restrict__ r_feat)
{
  __shared__ float dnn[9][64];
  __shared__ float dsv[64], rsv[64];
  __shared__ float disp[3][9];
  __shared__ float hbuf[2][9][Dh_];
  __shared__ float logit[2][9];
  __shared__ float attnw[2][9];

  int s = blockIdx.x;
  int b = s / Ns_, n = s - b * Ns_;
  int t = threadIdx.x;                 // channel
  int p = pc_idx[b * Ns_ + n];
  const bf16_t* d0b = d0 + (size_t)b * C_ * HW_;
  const bf16_t* r0b = r0 + (size_t)b * C_ * HW_;
  dsv[t] = (float)d0b[(size_t)t * HW_ + p];
  rsv[t] = (float)r0b[(size_t)t * HW_ + p];
  for (int k = 0; k < K_; ++k) {
    int pk = nbrs_idx[((size_t)b * Ns_ + n) * K_ + k];
    dnn[k][t] = (float)d0b[(size_t)t * HW_ + pk];
  }
  if (t < 27) {
    int dd = t / 9, k = t - dd * 9;
    disp[dd][k] = nbrs_disp[(((size_t)b * 3 + dd) * Ns_ + n) * K_ + k];
  }
  __syncthreads();

  // hidden layer: h[k][j] = leaky(feats[k] . w1[:,j] + b1[j]) for both MLPs
  for (int j = t; j < Dh_; j += 64) {
    const float* dr = dw1T + j * D_;
    const float* rr = rw1T + j * D_;
    for (int k = 0; k < K_; ++k) {
      float accd = db1[j], accr = rb1[j];
      for (int i = 0; i < 64; ++i) {
        float f = dnn[k][i] - dsv[i];
        accd += f * dr[i];
        accr += f * rr[i];
      }
      for (int i = 0; i < 64; ++i) {
        float f = dnn[k][i] - rsv[i];        // r_nn == d_nn (faithful)
        accd += f * dr[64 + i];
        accr += f * rr[64 + i];
      }
      for (int i = 0; i < 3; ++i) {
        float f = disp[i][k];
        accd += f * dr[128 + i];
        accr += f * rr[128 + i];
      }
      hbuf[0][k][j] = (accd > 0.f) ? accd : 0.2f * accd;
      hbuf[1][k][j] = (accr > 0.f) ? accr : 0.2f * accr;
    }
  }
  __syncthreads();
  if (t < 18) {
    int m = t / 9, k = t - m * 9;
    const float* w2 = m ? rw2 : dw2;
    float acc = m ? rb2[0] : db2[0];
    for (int j = 0; j < Dh_; ++j) acc += hbuf[m][k][j] * w2[j];
    logit[m][k] = acc;
  }
  __syncthreads();
  if (t < 2) {
    float mx = logit[t][0];
    for (int k = 1; k < K_; ++k) mx = fmaxf(mx, logit[t][k]);
    float e[9], sum = 0.f;
    for (int k = 0; k < K_; ++k) { e[k] = __expf(logit[t][k] - mx); sum += e[k]; }
    float inv = 1.f / sum;
    for (int k = 0; k < K_; ++k) attnw[t][k] = e[k] * inv;
  }
  __syncthreads();
  float accd = 0.f, accr = 0.f;
  #pragma unroll
  for (int k = 0; k < K_; ++k) {
    accd += attnw[0][k] * dnn[k][t];
    accr += attnw[1][k] * dnn[k][t];
  }
  d_feat[((size_t)b * C_ + t) * Ns_ + n] = accd + d_bias[t];
  r_feat[((size_t)b * C_ + t) * Ns_ + n] = accr + r_bias[t];
}

// ------------------------------- scatter -----------------------------------
// Faithful mask quirk: only channel 0 is replaced; channels 1..63 are ADDED.
__global__ void scatter_kernel(const int* __restrict__ pc_idx,
                               const float* __restrict__ d_feat, const float* __restrict__ r_feat,
                               bf16_t* __restrict__ d0, bf16_t* __restrict__ r0) {
  int i = blockIdx.x * blockDim.x + threadIdx.x;
  if (i >= B_ * Ns_ * C_) return;
  int c = i & 63;
  int n = (i >> 6) % Ns_;
  int b = i / (Ns_ * C_);
  int p = pc_idx[b * Ns_ + n];
  size_t o = ((size_t)b * C_ + c) * HW_ + p;
  size_t f = ((size_t)b * C_ + c) * Ns_ + n;
  float df = d_feat[f], rf = r_feat[f];
  float dv = (c == 0) ? df : (float)d0[o] + df;
  float rv = (c == 0) ? rf : (float)r0[o] + rf;
  d0[o] = (bf16_t)dv;
  r0[o] = (bf16_t)rv;
}

// ------------------------------- launcher ----------------------------------
extern "C" void kernel_launch(void* const* d_in, const int* in_sizes, int n_in,
                              void* d_out, int out_size, void* d_ws, size_t ws_size,
                              hipStream_t stream)
{
  (void)in_sizes; (void)n_in; (void)out_size; (void)ws_size;
  const float* rgb       = (const float*)d_in[0];
  const float* sdepth    = (const float*)d_in[1];
  const int*   pc_idx    = (const int*)  d_in[2];
  const int*   nbrs_idx  = (const int*)  d_in[3];
  const float* nbrs_disp = (const float*)d_in[4];
  const float* d_c0w = (const float*)d_in[5];  const float* d_c0b = (const float*)d_in[6];
  const float* d_c1w = (const float*)d_in[7];  const float* d_c1b = (const float*)d_in[8];
  const float* d_c2w = (const float*)d_in[9];  const float* d_c2b = (const float*)d_in[10];
  const float* r_c0w = (const float*)d_in[11]; const float* r_c0b = (const float*)d_in[12];
  const float* r_c1w = (const float*)d_in[13]; const float* r_c1b = (const float*)d_in[14];
  const float* r_c2w = (const float*)d_in[15]; const float* r_c2b = (const float*)d_in[16];
  const float* d_w1  = (const float*)d_in[17]; const float* d_b1  = (const float*)d_in[18];
  const float* d_w2  = (const float*)d_in[19]; const float* d_b2  = (const float*)d_in[20];
  const float* r_w1  = (const float*)d_in[21]; const float* r_b1  = (const float*)d_in[22];
  const float* r_w2  = (const float*)d_in[23]; const float* r_b2  = (const float*)d_in[24];
  const float* d_bias = (const float*)d_in[25];
  const float* r_bias = (const float*)d_in[26];

  const size_t NMAP = (size_t)B_ * C_ * HW_;   // 39,845,888 elements per map set
  char* ws = (char*)d_ws;
  size_t off = 0;
  auto walloc = [&](size_t bytes) -> void* {
    void* pp = (void*)(ws + off);
    off = (off + bytes + 255) & ~(size_t)255;
    return pp;
  };
  bf16_t* sdepth_bf = (bf16_t*)walloc(NMAP * 2);
  bf16_t* rgb_bf    = (bf16_t*)walloc(NMAP * 2);
  bf16_t* d0_bf     = (bf16_t*)walloc(NMAP * 2);
  bf16_t* r0_bf     = (bf16_t*)walloc(NMAP * 2);
  bf16_t* Wb        = (bf16_t*)walloc((size_t)6 * C_ * KDIM * 2);
  float*  d_feat    = (float*) walloc((size_t)B_ * C_ * Ns_ * 4);
  float*  r_feat    = (float*) walloc((size_t)B_ * C_ * Ns_ * 4);
  float*  dw1T      = (float*) walloc((size_t)D_ * Dh_ * 4);
  float*  rw1T      = (float*) walloc((size_t)D_ * Dh_ * 4);

  float* dout   = (float*)d_out;
  float* dout_d = dout;                // holds d1, then relu(d2+d1)
  float* dout_r = dout + NMAP;         // holds r1, then relu(r2+r1)

  { // weights -> bf16, K reordered to kpos*64+ci
    int n = 6 * C_ * KDIM;
    prep_weights_kernel<<<(n + 255) / 256, 256, 0, stream>>>(
        d_c0w, d_c1w, d_c2w, r_c0w, r_c1w, r_c2w, Wb);
  }
  { // MLP first-layer weight transpose
    int n = D_ * Dh_;
    prep_w1_kernel<<<(n + 255) / 256, 256, 0, stream>>>(d_w1, r_w1, dw1T, rw1T);
  }
  { // inputs -> bf16
    long n = (long)NMAP;
    int blocks = (int)((n + 255) / 256);
    cvt_kernel<<<blocks, 256, 0, stream>>>(sdepth, sdepth_bf, n);
    cvt_kernel<<<blocks, 256, 0, stream>>>(rgb,    rgb_bf,    n);
  }

  const int convGrid = B_ * H_ * NBLK; // 9728 blocks of 8 waves
  bf16_t* Wd0 = Wb + 0 * C_ * KDIM;
  bf16_t* Wd1 = Wb + 1 * C_ * KDIM;
  bf16_t* Wd2 = Wb + 2 * C_ * KDIM;
  bf16_t* Wr0 = Wb + 3 * C_ * KDIM;
  bf16_t* Wr1 = Wb + 4 * C_ * KDIM;
  bf16_t* Wr2 = Wb + 5 * C_ * KDIM;

  // d0 = relu(conv0(sdepth)) -> bf16; d1 = conv1(sdepth) -> d_out (f32 scratch)
  conv3x3_wmma<<<convGrid, 256, 0, stream>>>(sdepth_bf, Wd0, d_c0b, nullptr, d0_bf, nullptr, 1);
  conv3x3_wmma<<<convGrid, 256, 0, stream>>>(sdepth_bf, Wd1, d_c1b, dout_d,  nullptr, nullptr, 0);
  conv3x3_wmma<<<convGrid, 256, 0, stream>>>(rgb_bf,    Wr0, r_c0b, nullptr, r0_bf, nullptr, 1);
  conv3x3_wmma<<<convGrid, 256, 0, stream>>>(rgb_bf,    Wr1, r_c1b, dout_r,  nullptr, nullptr, 0);

  // attention -> d_feat / r_feat (reads pre-update d0/r0)
  attn_kernel<<<B_ * Ns_, 64, 0, stream>>>(d0_bf, r0_bf, pc_idx, nbrs_idx, nbrs_disp,
      dw1T, d_b1, d_w2, d_b2, rw1T, r_b1, r_w2, r_b2, d_bias, r_bias, d_feat, r_feat);

  { // scatter update into d0/r0 (unique indices -> race-free)
    int n = B_ * Ns_ * C_;
    scatter_kernel<<<(n + 255) / 256, 256, 0, stream>>>(pc_idx, d_feat, r_feat, d0_bf, r0_bf);
  }

  // final fused conv2 + residual + relu, in place on d_out
  conv3x3_wmma<<<convGrid, 256, 0, stream>>>(d0_bf, Wd2, d_c2b, dout_d, nullptr, dout_d, 1);
  conv3x3_wmma<<<convGrid, 256, 0, stream>>>(r0_bf, Wr2, r_c2b, dout_r, nullptr, dout_r, 1);
}
